// EdgeMLP_18983755448605
// MI455X (gfx1250) — matrix operand order repaired and checked by reading, not compiled, not run on hardware
//
#include <hip/hip_runtime.h>
#include <hip/hip_bf16.h>
#include <stdint.h>

#define EMB     64
#define HID     128
#define K2D     128        // 2*EMB
#define NGRAPH  512
#define GN_EPS  1e-5f

typedef unsigned int   u32;
typedef unsigned short u16;
typedef __bf16 bf16_t;
typedef bf16_t v16bf __attribute__((ext_vector_type(16)));
typedef float  v8f   __attribute__((ext_vector_type(8)));
typedef u32    u32x4 __attribute__((ext_vector_type(4)));

union FragAB { u32x4 u[2]; v16bf bf; };

__device__ __forceinline__ u16 f2bf(float f) {
  u32 u = __float_as_uint(f);
  u32 r = u + 0x7FFFu + ((u >> 16) & 1u);   // round-to-nearest-even
  return (u16)(r >> 16);
}

// ---------------------------------------------------------------- converts
__global__ void k_cvt_bf16(const float* __restrict__ src, u16* __restrict__ dst, int n) {
  for (int i = blockIdx.x * blockDim.x + threadIdx.x; i < n; i += gridDim.x * blockDim.x)
    dst[i] = f2bf(src[i]);
}

// ---------------------------------------------------------------- counting sort
__global__ void k_zero_counts(u32* counts) { counts[threadIdx.x] = 0u; }

__global__ void k_hist(const int* __restrict__ col, const int* __restrict__ batch,
                       u32* __restrict__ counts, int E) {
  __shared__ u32 lh[NGRAPH];
  for (int i = threadIdx.x; i < NGRAPH; i += blockDim.x) lh[i] = 0u;
  __syncthreads();
  for (int e = blockIdx.x * blockDim.x + threadIdx.x; e < E; e += gridDim.x * blockDim.x) {
    int g = batch[col[e]];
    atomicAdd(&lh[g], 1u);
  }
  __syncthreads();
  for (int i = threadIdx.x; i < NGRAPH; i += blockDim.x)
    if (lh[i]) atomicAdd(&counts[i], lh[i]);
}

__global__ void k_scan(const u32* __restrict__ counts, u32* __restrict__ offsets,
                       u32* __restrict__ cursor) {
  __shared__ u32 buf[NGRAPH];
  int t = threadIdx.x;
  u32 c = counts[t];
  buf[t] = c;
  __syncthreads();
  for (int d = 1; d < NGRAPH; d <<= 1) {
    u32 v = (t >= d) ? buf[t - d] : 0u;
    __syncthreads();
    buf[t] += v;
    __syncthreads();
  }
  u32 excl = buf[t] - c;
  offsets[t] = excl;
  cursor[t]  = excl;
}

__global__ void k_scatter(const int* __restrict__ col, const int* __restrict__ batch,
                          u32* __restrict__ cursor, u32* __restrict__ sortedE, int E) {
  for (int e = blockIdx.x * blockDim.x + threadIdx.x; e < E; e += gridDim.x * blockDim.x) {
    int g = batch[col[e]];
    u32 pos = atomicAdd(&cursor[g], 1u);
    sortedE[pos] = (u32)e;
  }
}

// ---------------------------------------------------------------- async gather
// Issue 8x GLOBAL_LOAD_ASYNC_TO_LDS_B128: each lane pulls its 128-byte half-row
// (64 bf16 of emb[node]) straight into this wave's LDS tile buffer (ASYNCcnt += 8).
// Invalid tail rows are clamped to a valid edge; they are masked out downstream.
__device__ __forceinline__ u32 issue_gather_async(const int* __restrict__ ei, int E,
                                                  const u16* __restrict__ embbf,
                                                  const u32* __restrict__ sortedE,
                                                  u32 base, u32 tb, u32 nE,
                                                  u16* sRow, int lane) {
  const int m = lane & 15, hsel = lane >> 4;
  const bool valid = (tb + (u32)m) < nE;
  const u32 idx = valid ? (base + tb + (u32)m) : base;
  const u32 eid = sortedE[idx];
  const int node = ei[(size_t)hsel * (size_t)E + (size_t)eid];
  const unsigned long long gaddr =
      (unsigned long long)(uintptr_t)(embbf + (size_t)node * EMB);
  const u32 laddr = (u32)(uintptr_t)(sRow + m * K2D + hsel * 64);
  asm volatile(
      "global_load_async_to_lds_b128 %0, %1, off\n\t"
      "global_load_async_to_lds_b128 %0, %1, off offset:16\n\t"
      "global_load_async_to_lds_b128 %0, %1, off offset:32\n\t"
      "global_load_async_to_lds_b128 %0, %1, off offset:48\n\t"
      "global_load_async_to_lds_b128 %0, %1, off offset:64\n\t"
      "global_load_async_to_lds_b128 %0, %1, off offset:80\n\t"
      "global_load_async_to_lds_b128 %0, %1, off offset:96\n\t"
      "global_load_async_to_lds_b128 %0, %1, off offset:112"
      :: "v"(laddr), "v"(gaddr) : "memory");
  return eid;
}

// ---------------------------------------------------------------- WMMA core
// 16(edges) x 128(hidden), K=128, via 4 k-steps x 8 n-tiles of v_wmma_f32_16x16x32_bf16.
__device__ __forceinline__ void mma_tile(const u16* __restrict__ sW1,   // [K=128][N=128] bf16
                                         const u16* __restrict__ sRow,  // [16][128] bf16
                                         const float* b1v, v8f acc[8], int lane) {
  const int m = lane & 15, hsel = lane >> 4;
#pragma unroll
  for (int n = 0; n < 8; ++n) {
#pragma unroll
    for (int i = 0; i < 8; ++i) acc[n][i] = b1v[n];
  }
#pragma unroll
  for (int s = 0; s < 4; ++s) {
    const int k0 = s * 32;
    FragAB a;  // A layout (16-bit 16x32): lanes<16 K=k0+{0..7,16..23}, lanes>=16 +8
    a.u[0] = *(const u32x4*)(sRow + m * K2D + k0 + hsel * 8);
    a.u[1] = *(const u32x4*)(sRow + m * K2D + k0 + 16 + hsel * 8);
    const u16* wcol = sW1 + (size_t)k0 * HID + m;  // B: lane = N column, K in same order as A
#pragma unroll
    for (int n = 0; n < 8; ++n) {
      FragAB b;
#pragma unroll
      for (int j = 0; j < 4; ++j) {
        const int ka = hsel * 8 + 2 * j;
        b.u[0][j] = (u32)wcol[ka * HID + n * 16] | ((u32)wcol[(ka + 1) * HID + n * 16] << 16);
        b.u[1][j] = (u32)wcol[(ka + 16) * HID + n * 16] | ((u32)wcol[(ka + 17) * HID + n * 16] << 16);
      }
      acc[n] = __builtin_amdgcn_wmma_f32_16x16x32_bf16(false, a.bf, false, b.bf,
                                                       (short)0, acc[n], false, false);
    }
  }
}

// ---------------------------------------------------------------- pass 1: per-graph stats
__global__ __launch_bounds__(256) void k_stats(
    const int* __restrict__ ei, const u16* __restrict__ embbf, const u16* __restrict__ W1bf,
    const float* __restrict__ b1, const float* __restrict__ gnw, const float* __restrict__ gnms,
    const u32* __restrict__ counts, const u32* __restrict__ offsets, const u32* __restrict__ sortedE,
    float* __restrict__ meanms, float* __restrict__ rstdw, int E) {
  __shared__ u16   sW1[HID * K2D];
  __shared__ u16   sF12[8][2][16 * K2D];    // double-buffered per wave
  __shared__ float sPart[8][2][HID];
  const int tid = threadIdx.x, lane = tid & 31, w = tid >> 5;
  { const u32x4* src = (const u32x4*)W1bf; u32x4* dst = (u32x4*)sW1;
    for (int i = tid; i < HID * K2D / 8; i += 256) dst[i] = src[i]; }
  __syncthreads();
  const int g = blockIdx.x;
  const u32 nE = counts[g], base = offsets[g];
  const u32 nTiles = (nE + 15u) >> 4;
  const int m = lane & 15, hsel = lane >> 4;
  float b1v[8];
#pragma unroll
  for (int n = 0; n < 8; ++n) b1v[n] = b1[n * 16 + m];
  float s1[8], s2[8];
#pragma unroll
  for (int n = 0; n < 8; ++n) { s1[n] = 0.f; s2[n] = 0.f; }
  u32 buf = 0;
  if ((u32)w < nTiles)
    issue_gather_async(ei, E, embbf, sortedE, base, ((u32)w) << 4, nE, &sF12[w][0][0], lane);
  for (u32 t = (u32)w; t < nTiles; t += 8u) {
    const u32 tb = t << 4;
    const bool hasNext = (t + 8u) < nTiles;
    if (hasNext)   // prefetch next tile into the other buffer while this one computes
      issue_gather_async(ei, E, embbf, sortedE, base, (t + 8u) << 4, nE,
                         &sF12[w][buf ^ 1u][0], lane);
    if (hasNext) asm volatile("s_wait_asynccnt 0x8" ::: "memory");
    else         asm volatile("s_wait_asynccnt 0x0" ::: "memory");
    v8f acc[8];
    mma_tile(sW1, &sF12[w][buf][0], b1v, acc, lane);
#pragma unroll
    for (int j = 0; j < 8; ++j) {
      const bool vr = (tb + (u32)(j + hsel * 8)) < nE;   // row M = j + 8*hsel
#pragma unroll
      for (int n = 0; n < 8; ++n) {
        float v = vr ? acc[n][j] : 0.f;
        s1[n] += v; s2[n] += v * v;
      }
    }
    buf ^= 1u;
  }
#pragma unroll
  for (int n = 0; n < 8; ++n) {     // combine half-waves (rows 0-7 with rows 8-15)
    s1[n] += __shfl_xor(s1[n], 16, 32);
    s2[n] += __shfl_xor(s2[n], 16, 32);
  }
  if (hsel == 0) {
#pragma unroll
    for (int n = 0; n < 8; ++n) { sPart[w][0][n * 16 + m] = s1[n]; sPart[w][1][n * 16 + m] = s2[n]; }
  }
  __syncthreads();
  if (tid < HID) {
    float su = 0.f, sq = 0.f;
    for (int ww = 0; ww < 8; ++ww) { su += sPart[ww][0][tid]; sq += sPart[ww][1][tid]; }
    const float cnt  = (float)(nE ? nE : 1u);
    const float mean = su / cnt;
    const float m2   = mean * gnms[tid];
    // E[(h - m*ms)^2] = E[h^2] - 2*m*ms*E[h] + (m*ms)^2
    const float var  = sq / cnt - 2.f * m2 * mean + m2 * m2;
    meanms[g * HID + tid] = m2;
    rstdw[g * HID + tid]  = gnw[tid] / sqrtf(var + GN_EPS);
  }
}

// ---------------------------------------------------------------- pass 2: normalize + W2 dot
__global__ __launch_bounds__(256) void k_out(
    const int* __restrict__ ei, const u16* __restrict__ embbf, const u16* __restrict__ W1bf,
    const float* __restrict__ b1, const float* __restrict__ gnb,
    const float* __restrict__ W2, const float* __restrict__ b2,
    const u32* __restrict__ counts, const u32* __restrict__ offsets, const u32* __restrict__ sortedE,
    const float* __restrict__ meanms, const float* __restrict__ rstdw,
    float* __restrict__ out, int E) {
  __shared__ u16 sW1[HID * K2D];
  __shared__ u16 sF12[8][2][16 * K2D];      // double-buffered per wave
  __shared__ u32 sEid[8][16];
  const int tid = threadIdx.x, lane = tid & 31, w = tid >> 5;
  { const u32x4* src = (const u32x4*)W1bf; u32x4* dst = (u32x4*)sW1;
    for (int i = tid; i < HID * K2D / 8; i += 256) dst[i] = src[i]; }
  __syncthreads();
  const int g = blockIdx.x;
  const u32 nE = counts[g], base = offsets[g];
  const u32 nTiles = (nE + 15u) >> 4;
  const int m = lane & 15, hsel = lane >> 4;
  float b1v[8], m2v[8], rsv[8], bv[8], w2v[8];
#pragma unroll
  for (int n = 0; n < 8; ++n) {
    const int N = n * 16 + m;
    b1v[n] = b1[N];
    m2v[n] = meanms[g * HID + N];
    rsv[n] = rstdw[g * HID + N];
    bv[n]  = gnb[N];
    w2v[n] = W2[N];
  }
  const float bias2 = b2[0];
  u32 buf = 0;
  u32 eid_cur = 0;
  if ((u32)w < nTiles)
    eid_cur = issue_gather_async(ei, E, embbf, sortedE, base, ((u32)w) << 4, nE,
                                 &sF12[w][0][0], lane);
  for (u32 t = (u32)w; t < nTiles; t += 8u) {
    const u32 tb = t << 4;
    const bool hasNext = (t + 8u) < nTiles;
    u32 eid_next = 0;
    if (hasNext)
      eid_next = issue_gather_async(ei, E, embbf, sortedE, base, (t + 8u) << 4, nE,
                                    &sF12[w][buf ^ 1u][0], lane);
    if (hasNext) asm volatile("s_wait_asynccnt 0x8" ::: "memory");
    else         asm volatile("s_wait_asynccnt 0x0" ::: "memory");
    if (hsel == 0) sEid[w][m] = eid_cur;
    v8f acc[8];
    mma_tile(sW1, &sF12[w][buf][0], b1v, acc, lane);
    float tj[8];
#pragma unroll
    for (int j = 0; j < 8; ++j) {
      float a2 = 0.f;
#pragma unroll
      for (int n = 0; n < 8; ++n) {
        float v = (acc[n][j] - m2v[n]) * rsv[n] + bv[n];
        v = v > 0.f ? v : 0.f;
        a2 += v * w2v[n];
      }
      tj[j] = a2;
    }
#pragma unroll
    for (int j = 0; j < 8; ++j) {   // reduce across the 16 N-lanes of each half-wave
#pragma unroll
      for (int d = 1; d < 16; d <<= 1) tj[j] += __shfl_xor(tj[j], d, 32);
    }
    if (m < 8) {                    // lanes 0-7 write rows 0-7, lanes 16-23 write rows 8-15
      const int M = m + hsel * 8;
      if (tb + (u32)M < nE) out[sEid[w][M]] = tj[m] + bias2;
    }
    eid_cur = eid_next;
    buf ^= 1u;
  }
}

// ---------------------------------------------------------------- host
extern "C" void kernel_launch(void* const* d_in, const int* in_sizes, int n_in,
                              void* d_out, int out_size, void* d_ws, size_t ws_size,
                              hipStream_t stream) {
  const float* emb   = (const float*)d_in[0];
  const float* W1    = (const float*)d_in[1];
  const float* b1    = (const float*)d_in[2];
  const float* gnw   = (const float*)d_in[3];
  const float* gnb   = (const float*)d_in[4];
  const float* gnms  = (const float*)d_in[5];
  const float* W2    = (const float*)d_in[6];
  const float* b2    = (const float*)d_in[7];
  const int*   ei    = (const int*)d_in[8];   // [2,E]: first E = col, next E = row
  const int*   batch = (const int*)d_in[9];
  float* out = (float*)d_out;
  const int E      = in_sizes[8] / 2;
  const int nNodes = in_sizes[0] / EMB;

  char* p = (char*)d_ws;
  auto carve = [&](size_t bytes) { char* r = p; p += (bytes + 255) & ~(size_t)255; return r; };
  u32*   counts  = (u32*)carve((size_t)NGRAPH * 4);
  u32*   cursor  = (u32*)carve((size_t)NGRAPH * 4);
  u32*   offsets = (u32*)carve((size_t)NGRAPH * 4);
  float* meanms  = (float*)carve((size_t)NGRAPH * HID * 4);
  float* rstdw   = (float*)carve((size_t)NGRAPH * HID * 4);
  u32*   sortedE = (u32*)carve((size_t)E * 4);
  u16*   W1bf    = (u16*)carve((size_t)HID * K2D * 2);
  u16*   embbf   = (u16*)carve((size_t)nNodes * EMB * 2);

  k_cvt_bf16<<<2048, 256, 0, stream>>>(emb, embbf, nNodes * EMB);
  k_cvt_bf16<<<64, 256, 0, stream>>>(W1, W1bf, HID * K2D);
  k_zero_counts<<<1, NGRAPH, 0, stream>>>(counts);
  k_hist<<<1024, 256, 0, stream>>>(ei, batch, counts, E);
  k_scan<<<1, NGRAPH, 0, stream>>>(counts, offsets, cursor);
  k_scatter<<<1024, 256, 0, stream>>>(ei, batch, cursor, sortedE, E);
  k_stats<<<NGRAPH, 256, 0, stream>>>(ei, embbf, W1bf, b1, gnw, gnms,
                                      counts, offsets, sortedE, meanms, rstdw, E);
  k_out<<<NGRAPH, 256, 0, stream>>>(ei, embbf, W1bf, b1, gnb, W2, b2,
                                    counts, offsets, sortedE, meanms, rstdw, out, E);
}
